// UniPhyBlock_31705448579412
// MI455X (gfx1250) — matrix operand order repaired and checked by reading, not compiled, not run on hardware
//
#include <hip/hip_runtime.h>
#include <hip/hip_bf16.h>

// ---------------------------------------------------------------------------
// UniPhyBlock for MI455X (gfx1250, wave32).  All GEMM-shaped work (3x3 conv
// implicit GEMM, complex encode/gate/decode, MoE FFN) runs on
// v_wmma_f32_16x16x32_bf16 (bf16 in, f32 accumulate).
// ---------------------------------------------------------------------------

typedef __bf16 bf16;
typedef __attribute__((ext_vector_type(16))) __bf16 v16bf;
typedef __attribute__((ext_vector_type(8)))  __bf16 v8bf;
typedef __attribute__((ext_vector_type(8)))  float  v8f;

#define Bn   2
#define Tn   16
#define Dn   64
#define Hn   48
#define Wn   48
#define D2   128
#define En   4
#define Fn   256
#define HW   (Hn * Wn)              // 2304
#define NTOK (Bn * Tn * HW)         // 73728
#define NIMG (Bn * Tn)              // 32
#define TILES_PER_IMG (HW / 128)    // 18
#define NBLK (NIMG * TILES_PER_IMG) // 576

#define WMMA_BF16(a, b, c) \
  __builtin_amdgcn_wmma_f32_16x16x32_bf16(false, (a), false, (b), (short)0, (c), false, false)

// ---- WMMA fragment loaders (layouts per CDNA5 ISA 7.12.2) ------------------
// A matrix 16x32 bf16, tile stored row-major [m][k] (stride in elements).
__device__ __forceinline__ v16bf frag_a16(const bf16* base, int stride,
                                          int m0, int kk, int lane) {
  const bf16* p = base + (m0 + (lane & 15)) * stride + kk + ((lane & 16) ? 8 : 0);
  union { v16bf v; v8bf h[2]; } u;
  u.h[0] = *(const v8bf*)(p);
  u.h[1] = *(const v8bf*)(p + 16);
  return u.v;
}
// B matrix 32x16 bf16, tile stored TRANSPOSED [n][k] (stride in elements).
__device__ __forceinline__ v16bf frag_b16(const bf16* base, int stride,
                                          int n0, int kk, int lane) {
  const bf16* p = base + (n0 + (lane & 15)) * stride + kk + ((lane & 16) ? 8 : 0);
  union { v16bf v; v8bf h[2]; } u;
  u.h[0] = *(const v8bf*)(p);
  u.h[1] = *(const v8bf*)(p + 16);
  return u.v;
}

// ---------------------------------------------------------------------------
// K0: build combined bf16 weights for the temporal WMMA GEMMs.
//  wenc_t[n][k] (128x128): [xr‖xi] @ Wenc -> [xe_re‖xe_im]
//  wgt_t [n][k] (64x64)  : transposed gate_w
//  wdec_t[n][k] (64x128) : [hr‖hi] @ Wdec -> Re(h @ dec)
// ---------------------------------------------------------------------------
__global__ __launch_bounds__(256) void k0_prep(
    const float* __restrict__ enc_re, const float* __restrict__ enc_im,
    const float* __restrict__ gate_w,
    const float* __restrict__ dec_re, const float* __restrict__ dec_im,
    bf16* __restrict__ wenc_t, bf16* __restrict__ wgt_t, bf16* __restrict__ wdec_t) {
  int tid = blockIdx.x * blockDim.x + threadIdx.x;
  if (tid < 128 * 128) {
    int n = tid >> 7, k = tid & 127;
    float v;
    if (n < 64) v = (k < 64) ? enc_re[k * 64 + n] : -enc_im[(k - 64) * 64 + n];
    else { int j = n - 64; v = (k < 64) ? enc_im[k * 64 + j] : enc_re[(k - 64) * 64 + j]; }
    wenc_t[n * 128 + k] = (bf16)v;
  } else if (tid < 128 * 128 + 64 * 64) {
    int i = tid - 128 * 128; int n = i >> 6, k = i & 63;
    wgt_t[n * 64 + k] = (bf16)gate_w[k * 64 + n];
  } else if (tid < 128 * 128 + 64 * 64 + 64 * 128) {
    int i = tid - (128 * 128 + 64 * 64); int n = i >> 7, k = i & 127;
    float v = (k < 64) ? dec_re[k * 64 + n] : -dec_im[(k - 64) * 64 + n];
    wdec_t[n * 128 + k] = (bf16)v;
  }
}

// ---------------------------------------------------------------------------
// K1: spatial complex layernorm -> bf16 token-major conv input xcat[token][c]
// ---------------------------------------------------------------------------
__global__ __launch_bounds__(128) void k1_spatial_norm(
    const float* __restrict__ xre, const float* __restrict__ xim,
    const float* __restrict__ g, const float* __restrict__ bb,
    bf16* __restrict__ xcat) {
  int token = blockIdx.x;
  int c = threadIdx.x;
  int bt = token / HW, pix = token % HW;
  float v = (c < Dn) ? xre[(bt * Dn + c) * HW + pix]
                     : xim[(bt * Dn + (c - Dn)) * HW + pix];
  __shared__ float red[128];
  red[c] = v; __syncthreads();
  for (int s = 64; s > 0; s >>= 1) { if (c < s) red[c] += red[c + s]; __syncthreads(); }
  float mean = red[0] * (1.0f / 128.0f);
  __syncthreads();
  float d = v - mean;
  red[c] = d * d; __syncthreads();
  for (int s = 64; s > 0; s >>= 1) { if (c < s) red[c] += red[c + s]; __syncthreads(); }
  float y = d * rsqrtf(red[0] * (1.0f / 128.0f) + 1e-5f) * g[c] + bb[c];
  xcat[token * 128 + c] = (bf16)y;
}

// ---------------------------------------------------------------------------
// K2: 3x3 SAME conv as implicit GEMM with WMMA (M=128 tok, N=128 oc, K=1152).
// ---------------------------------------------------------------------------
#define SA2 40
__global__ __launch_bounds__(256) void k2_conv(
    const bf16* __restrict__ xcat, const float* __restrict__ convw,
    const float* __restrict__ convb,
    const float* __restrict__ xre, const float* __restrict__ xim,
    float* __restrict__ xsp) {
  __shared__ __align__(16) bf16 As[128 * SA2];
  __shared__ __align__(16) bf16 Bs[128 * SA2];
  int tid = threadIdx.x, lane = tid & 31, wave = tid >> 5;
  int img = blockIdx.x / TILES_PER_IMG;
  int pix0 = (blockIdx.x % TILES_PER_IMG) * 128;

  v8f acc[8];
  for (int i = 0; i < 8; ++i) for (int v = 0; v < 8; ++v) acc[i][v] = 0.0f;

  for (int r = 0; r < 9; ++r) {
    int ky = r / 3 - 1, kx = r % 3 - 1;
    for (int ic0 = 0; ic0 < 128; ic0 += 32) {
      __syncthreads();
      for (int g8 = tid; g8 < 512; g8 += 256) {
        int m = g8 >> 2, kc0 = (g8 & 3) * 8;
        int pix = pix0 + m;
        int hh = pix / Wn + ky, ww = pix % Wn + kx;
        v8bf val;
        if (hh >= 0 && hh < Hn && ww >= 0 && ww < Wn) {
          val = *(const v8bf*)&xcat[((size_t)(img * HW + hh * Wn + ww)) * 128 + ic0 + kc0];
        } else {
          for (int j = 0; j < 8; ++j) val[j] = (bf16)0.0f;
        }
        *(v8bf*)&As[m * SA2 + kc0] = val;
      }
      for (int idx = tid; idx < 4096; idx += 256) {
        int n = idx >> 5, kc = idx & 31;
        Bs[n * SA2 + kc] = (bf16)convw[n * 1152 + (ic0 + kc) * 9 + r];
      }
      __syncthreads();
      v16bf a = frag_a16(As, SA2, wave * 16, 0, lane);
      v16bf bm[8];
      for (int nt = 0; nt < 8; ++nt) bm[nt] = frag_b16(Bs, SA2, nt * 16, 0, lane);
      for (int nt = 0; nt < 8; ++nt) acc[nt] = WMMA_BF16(a, bm[nt], acc[nt]);
    }
  }
  int nl = lane & 15, mh = (lane >> 4) * 8;
  for (int nt = 0; nt < 8; ++nt) {
    int oc = nt * 16 + nl;
    float bias = convb[oc];
    for (int v = 0; v < 8; ++v) {
      int pix = pix0 + wave * 16 + mh + v;
      float xo = (oc < Dn) ? xre[(img * Dn + oc) * HW + pix]
                           : xim[(img * Dn + (oc - Dn)) * HW + pix];
      xsp[((size_t)(img * HW + pix)) * 128 + oc] = acc[nt][v] + bias + xo;
    }
  }
}

// ---------------------------------------------------------------------------
// K3: fused temporal path, all matmuls on WMMA.  Block = 64 pixels (M=64),
// 128 threads (4 waves, wave = 16-pixel m-tile), loops t=0..15 sequentially:
//   LN -> enc GEMM(K=128,N=128) -> gate GEMM(K=64,N=64) -> elementwise SSM
//   scan (f32 state in registers) -> dec GEMM(K=128,N=64) -> fl.
// Weight B-fragments are loaded straight from global bf16 (L2-resident).
// ---------------------------------------------------------------------------
__global__ __launch_bounds__(128) void k3_temporal(
    const float* __restrict__ xsp, const float* __restrict__ dt,
    const float* __restrict__ tg, const float* __restrict__ tb,
    const bf16* __restrict__ wenc_t, const bf16* __restrict__ wgt_t,
    const bf16* __restrict__ wdec_t,
    const float* __restrict__ gate_b,
    const float* __restrict__ lam_re, const float* __restrict__ lam_im,
    const float* __restrict__ src_re, const float* __restrict__ src_im,
    const float* __restrict__ noise_re, const float* __restrict__ noise_im,
    float* __restrict__ fl) {
  __shared__ __align__(16) bf16 XT[64 * 128];   // LN result, later h-state tile
  __shared__ __align__(16) bf16 XER[64 * 64];   // x_enc.real (gate GEMM input)
  __shared__ float psum[128], psq[128];
  int tid = threadIdx.x, lane = tid & 31, wave = tid >> 5;
  int b = blockIdx.x / 36;
  int pix0 = (blockIdx.x % 36) * 64;
  int nl = lane & 15, mh = (lane >> 4) * 8;

  // lambda per owned channel j = nt*16+nl
  float lre[4], lim[4], den[4], sre[4], sim[4], gbv[4];
  for (int nt = 0; nt < 4; ++nt) {
    int j = nt * 16 + nl;
    float lr = lam_re[j];
    lre[nt] = -((lr > 20.f) ? lr : log1pf(expf(lr)));
    lim[nt] = lam_im[j];
    den[nt] = lre[nt] * lre[nt] + lim[nt] * lim[nt];
    sre[nt] = src_re[j]; sim[nt] = src_im[j]; gbv[nt] = gate_b[j];
  }
  float hre[4][8], him[4][8];
  for (int nt = 0; nt < 4; ++nt)
    for (int v = 0; v < 8; ++v) { hre[nt][v] = 0.f; him[nt][v] = 0.f; }

  for (int t = 0; t < Tn; ++t) {
    size_t tokbase = (size_t)(b * Tn + t) * HW + pix0;
    // ---- LN into XT: 2 threads per token, 64 channels each ----
    {
      int m = tid >> 1, hs = tid & 1;
      const float* xr = &xsp[(tokbase + m) * 128 + hs * 64];
      float s = 0.f, sq = 0.f;
      for (int c2 = 0; c2 < 64; ++c2) { float v = xr[c2]; s += v; sq += v * v; }
      psum[tid] = s; psq[tid] = sq;
      __syncthreads();
      float st = psum[tid ^ 1] + s, sqt = psq[tid ^ 1] + sq;
      float mean = st * (1.f / 128.f);
      float rstd = rsqrtf(sqt * (1.f / 128.f) - mean * mean + 1e-5f);
      for (int c2 = 0; c2 < 64; ++c2) {
        int c = hs * 64 + c2;
        XT[m * 128 + c] = (bf16)((xr[c2] - mean) * rstd * tg[c] + tb[c]);
      }
    }
    __syncthreads();
    // ---- enc GEMM: [64x128] @ wenc_t -> acc_e[8] (re: nt 0..3, im: 4..7) ----
    v8f acc_e[8];
    for (int i = 0; i < 8; ++i) for (int v = 0; v < 8; ++v) acc_e[i][v] = 0.f;
    for (int kk = 0; kk < 128; kk += 32) {
      v16bf a = frag_a16(XT, 128, wave * 16, kk, lane);
      v16bf bm[8];
      for (int nt = 0; nt < 8; ++nt) bm[nt] = frag_b16(wenc_t, 128, nt * 16, kk, lane);
      for (int nt = 0; nt < 8; ++nt) acc_e[nt] = WMMA_BF16(a, bm[nt], acc_e[nt]);
    }
    // stash x_enc.real as bf16 for the gate GEMM
    for (int nt = 0; nt < 4; ++nt)
      for (int v = 0; v < 8; ++v)
        XER[(wave * 16 + mh + v) * 64 + nt * 16 + nl] = (bf16)acc_e[nt][v];
    __syncthreads();
    // ---- gate GEMM: [64x64] @ wgt_t -> acc_g[4] ----
    v8f acc_g[4];
    for (int i = 0; i < 4; ++i) for (int v = 0; v < 8; ++v) acc_g[i][v] = 0.f;
    for (int kk = 0; kk < 64; kk += 32) {
      v16bf a = frag_a16(XER, 64, wave * 16, kk, lane);
      v16bf bm[4];
      for (int nt = 0; nt < 4; ++nt) bm[nt] = frag_b16(wgt_t, 64, nt * 16, kk, lane);
      for (int nt = 0; nt < 4; ++nt) acc_g[nt] = WMMA_BF16(a, bm[nt], acc_g[nt]);
    }
    __syncthreads();  // enc reads of XT done; safe to overwrite with h
    // ---- elementwise: silu gate, forcing, noise, scan step; h -> XT ----
    float dtv = dt[b * Tn + t];
    float nsc = 0.01f * sqrtf(dtv);
    for (int nt = 0; nt < 4; ++nt) {
      int j = nt * 16 + nl;
      float ex = expf(lre[nt] * dtv);
      float ar = ex * cosf(lim[nt] * dtv), ai = ex * sinf(lim[nt] * dtv);
      float fr = ((ar - 1.f) * lre[nt] + ai * lim[nt]) / den[nt];
      float fi = (ai * lre[nt] - (ar - 1.f) * lim[nt]) / den[nt];
      for (int v = 0; v < 8; ++v) {
        int m = wave * 16 + mh + v;
        float gp = acc_g[nt][v] + gbv[nt];
        float gate = gp / (1.f + expf(-gp));
        float ur = acc_e[nt][v] * gate + sre[nt];
        float ui = acc_e[nt + 4][v] * gate + sim[nt];
        size_t nidx = ((size_t)(b * HW + pix0 + m) * Tn + t) * 64 + j;
        float uur = ur * fr - ui * fi + nsc * noise_re[nidx];
        float uui = ur * fi + ui * fr + nsc * noise_im[nidx];
        float h2r = ar * hre[nt][v] - ai * him[nt][v] + uur;
        float h2i = ar * him[nt][v] + ai * hre[nt][v] + uui;
        hre[nt][v] = h2r; him[nt][v] = h2i;
        XT[m * 128 + j]      = (bf16)h2r;
        XT[m * 128 + 64 + j] = (bf16)h2i;
      }
    }
    __syncthreads();
    // ---- dec GEMM: [64x128] @ wdec_t -> drift acc_d[4] ----
    v8f acc_d[4];
    for (int i = 0; i < 4; ++i) for (int v = 0; v < 8; ++v) acc_d[i][v] = 0.f;
    for (int kk = 0; kk < 128; kk += 32) {
      v16bf a = frag_a16(XT, 128, wave * 16, kk, lane);
      v16bf bm[4];
      for (int nt = 0; nt < 4; ++nt) bm[nt] = frag_b16(wdec_t, 128, nt * 16, kk, lane);
      for (int nt = 0; nt < 4; ++nt) acc_d[nt] = WMMA_BF16(a, bm[nt], acc_d[nt]);
    }
    // ---- fl = x_spatial + drift (real), imag passthrough ----
    for (int nt = 0; nt < 4; ++nt) {
      int c = nt * 16 + nl;
      for (int v = 0; v < 8; ++v) {
        size_t tok = tokbase + wave * 16 + mh + v;
        fl[tok * 128 + c]      = xsp[tok * 128 + c] + acc_d[nt][v];
        fl[tok * 128 + 64 + c] = xsp[tok * 128 + 64 + c];
      }
    }
    __syncthreads();  // before next-t LN overwrites XT
  }
}

// ---------------------------------------------------------------------------
// K4: probs[bt][e] = softmax(mean_{h,w}(fl) @ moe_gate_w)
// ---------------------------------------------------------------------------
__global__ __launch_bounds__(128) void k4_probs(
    const float* __restrict__ fl, const float* __restrict__ mg,
    float* __restrict__ probs) {
  int bt = blockIdx.x, c = threadIdx.x;
  float s = 0.f;
  const float* base = &fl[(size_t)bt * HW * 128 + c];
  for (int p = 0; p < HW; ++p) s += base[(size_t)p * 128];
  __shared__ float mean[128];
  __shared__ float logit[4];
  mean[c] = s * (1.f / HW);
  __syncthreads();
  if (c < 4) {
    float l = 0.f;
    for (int d = 0; d < 128; ++d) l += mean[d] * mg[d * 4 + c];
    logit[c] = l;
  }
  __syncthreads();
  if (c == 0) {
    float mx = fmaxf(fmaxf(logit[0], logit[1]), fmaxf(logit[2], logit[3]));
    float e0 = expf(logit[0] - mx), e1 = expf(logit[1] - mx);
    float e2 = expf(logit[2] - mx), e3 = expf(logit[3] - mx);
    float dd = e0 + e1 + e2 + e3;
    probs[bt * 4 + 0] = e0 / dd; probs[bt * 4 + 1] = e1 / dd;
    probs[bt * 4 + 2] = e2 / dd; probs[bt * 4 + 3] = e3 / dd;
  }
}

// ---------------------------------------------------------------------------
// K5: MoE FFN with WMMA (FL tile reused across 4 experts; gelu*p_e folded).
// ---------------------------------------------------------------------------
#define SFL 128
#define SW1 40
#define SH1 64
#define SW2 40
__global__ __launch_bounds__(256) void k5_moe(
    const float* __restrict__ fl, const float* __restrict__ probs,
    const float* __restrict__ w1, const float* __restrict__ b1,
    const float* __restrict__ w2, const float* __restrict__ b2,
    float* __restrict__ out) {
  __shared__ __align__(16) bf16 FLs[128 * SFL];
  __shared__ __align__(16) bf16 W1s[64 * SW1];
  __shared__ __align__(16) bf16 H1s[128 * SH1];
  __shared__ __align__(16) bf16 W2s[128 * SW2];
  int tid = threadIdx.x, lane = tid & 31, wave = tid >> 5;
  int bt = blockIdx.x / TILES_PER_IMG;
  int tok0 = bt * HW + (blockIdx.x % TILES_PER_IMG) * 128;

  for (int idx = tid; idx < 128 * 128; idx += 256) {
    int m = idx >> 7, c = idx & 127;
    FLs[m * SFL + c] = (bf16)fl[((size_t)(tok0 + m)) * 128 + c];
  }
  v8f acc2[8];
  for (int i = 0; i < 8; ++i) for (int v = 0; v < 8; ++v) acc2[i][v] = 0.f;

  for (int e = 0; e < En; ++e) {
    float pe = probs[bt * 4 + e];
    for (int fc = 0; fc < 4; ++fc) {
      int f0 = fc * 64;
      v8f acc1[4];
      for (int i = 0; i < 4; ++i) for (int v = 0; v < 8; ++v) acc1[i][v] = 0.f;
      for (int kk = 0; kk < 128; kk += 32) {
        __syncthreads();
        for (int idx = tid; idx < 64 * 32; idx += 256) {
          int n = idx >> 5, kc = idx & 31;
          W1s[n * SW1 + kc] = (bf16)w1[((size_t)(e * 128 + kk + kc)) * 256 + f0 + n];
        }
        __syncthreads();
        v16bf a = frag_a16(FLs, SFL, wave * 16, kk, lane);
        v16bf bm[4];
        for (int nt = 0; nt < 4; ++nt) bm[nt] = frag_b16(W1s, SW1, nt * 16, 0, lane);
        for (int nt = 0; nt < 4; ++nt) acc1[nt] = WMMA_BF16(a, bm[nt], acc1[nt]);
      }
      __syncthreads();
      int nl = lane & 15, mh = (lane >> 4) * 8;
      for (int nt = 0; nt < 4; ++nt) {
        int fidx = f0 + nt * 16 + nl;
        float bias = b1[e * 256 + fidx];
        for (int v = 0; v < 8; ++v) {
          float x = acc1[nt][v] + bias;
          float gl = 0.5f * x * (1.f + tanhf(0.7978845608f * (x + 0.044715f * x * x * x)));
          int m = wave * 16 + mh + v;
          H1s[m * SH1 + nt * 16 + nl] = (bf16)(pe * gl);
        }
      }
      __syncthreads();
      for (int kk2 = 0; kk2 < 64; kk2 += 32) {
        for (int idx = tid; idx < 128 * 32; idx += 256) {
          int n = idx >> 5, kc = idx & 31;
          W2s[n * SW2 + kc] = (bf16)w2[((size_t)(e * 256 + f0 + kk2 + kc)) * 128 + n];
        }
        __syncthreads();
        v16bf a2 = frag_a16(H1s, SH1, wave * 16, kk2, lane);
        v16bf bm[8];
        for (int nt = 0; nt < 8; ++nt) bm[nt] = frag_b16(W2s, SW2, nt * 16, 0, lane);
        for (int nt = 0; nt < 8; ++nt) acc2[nt] = WMMA_BF16(a2, bm[nt], acc2[nt]);
        __syncthreads();
      }
    }
  }
  int nl = lane & 15, mh = (lane >> 4) * 8;
  for (int nt = 0; nt < 8; ++nt) {
    int c = nt * 16 + nl;
    float bsum = 0.f;
    for (int e = 0; e < En; ++e) bsum += probs[bt * 4 + e] * b2[e * 128 + c];
    for (int v = 0; v < 8; ++v) {
      int token = tok0 + wave * 16 + mh + v;
      int pix = token % HW;
      float val = acc2[nt][v] + bsum + fl[(size_t)token * 128 + c];
      out[((size_t)bt * 128 + c) * HW + pix] = val;
    }
  }
}

// ---------------------------------------------------------------------------
extern "C" void kernel_launch(void* const* d_in, const int* in_sizes, int n_in,
                              void* d_out, int out_size, void* d_ws, size_t ws_size,
                              hipStream_t stream) {
  const float* x_re    = (const float*)d_in[0];
  const float* x_im    = (const float*)d_in[1];
  const float* dt      = (const float*)d_in[2];
  const float* ln_s_g  = (const float*)d_in[3];
  const float* ln_s_b  = (const float*)d_in[4];
  const float* ln_t_g  = (const float*)d_in[5];
  const float* ln_t_b  = (const float*)d_in[6];
  const float* conv_w  = (const float*)d_in[7];
  const float* conv_b  = (const float*)d_in[8];
  const float* enc_re  = (const float*)d_in[9];
  const float* enc_im  = (const float*)d_in[10];
  const float* dec_re  = (const float*)d_in[11];
  const float* dec_im  = (const float*)d_in[12];
  const float* gate_w  = (const float*)d_in[13];
  const float* gate_b  = (const float*)d_in[14];
  const float* lam_re  = (const float*)d_in[15];
  const float* lam_im  = (const float*)d_in[16];
  const float* src_re  = (const float*)d_in[17];
  const float* src_im  = (const float*)d_in[18];
  const float* noise_re= (const float*)d_in[19];
  const float* noise_im= (const float*)d_in[20];
  const float* moe_gw  = (const float*)d_in[21];
  const float* w1      = (const float*)d_in[22];
  const float* b1      = (const float*)d_in[23];
  const float* w2      = (const float*)d_in[24];
  const float* b2      = (const float*)d_in[25];
  float* out = (float*)d_out;

  // workspace layout (16B-aligned sections)
  char* ws = (char*)d_ws;
  const size_t TOKBYTES_BF16 = (size_t)NTOK * 128 * sizeof(bf16);   // 18.9 MB
  const size_t TOKBYTES_F32  = (size_t)NTOK * 128 * sizeof(float);  // 37.7 MB
  bf16*  xcatb = (bf16*)ws;
  float* xsp   = (float*)(ws + TOKBYTES_BF16);
  float* flv   = (float*)(ws + TOKBYTES_BF16 + TOKBYTES_F32);
  char*  tail  = ws + TOKBYTES_BF16 + 2 * TOKBYTES_F32;
  float* probs = (float*)tail;                       // 512 B
  bf16*  wenc_t = (bf16*)(tail + 512);               // 32768 B
  bf16*  wgt_t  = (bf16*)(tail + 512 + 32768);       //  8192 B
  bf16*  wdec_t = (bf16*)(tail + 512 + 32768 + 8192);// 16384 B

  k0_prep<<<112, 256, 0, stream>>>(enc_re, enc_im, gate_w, dec_re, dec_im,
                                   wenc_t, wgt_t, wdec_t);
  k1_spatial_norm<<<NTOK, 128, 0, stream>>>(x_re, x_im, ln_s_g, ln_s_b, xcatb);
  k2_conv<<<NBLK, 256, 0, stream>>>(xcatb, conv_w, conv_b, x_re, x_im, xsp);
  k3_temporal<<<Bn * HW / 64, 128, 0, stream>>>(xsp, dt, ln_t_g, ln_t_b,
                                                wenc_t, wgt_t, wdec_t,
                                                gate_b, lam_re, lam_im,
                                                src_re, src_im,
                                                noise_re, noise_im, flv);
  k4_probs<<<NIMG, 128, 0, stream>>>(flv, moe_gw, probs);
  k5_moe<<<NBLK, 256, 0, stream>>>(flv, probs, w1, b1, w2, b2, out);
}
